// SepConv2d_5162550690024
// MI455X (gfx1250) — matrix operand order repaired and checked by reading, not compiled, not run on hardware
//
#include <hip/hip_runtime.h>
#include <hip/hip_bf16.h>
#include <stdint.h>

// ---- problem constants (from reference) ----
#define BB   8
#define CC   128
#define HH   64
#define WW   64
#define HWXY 4096            // H*W
#define CO1  216             // offset-conv out channels
#define CO1P 224             // padded to multiple of 16
#define K1   1152            // C*3*3
#define GG   8
#define CGRP 16              // C / G
#define KTAP 9
#define LPAD 40              // LDS row stride in bf16: 80B = 20 banks, 16B aligned

typedef __attribute__((ext_vector_type(16))) __bf16 v16bf;
typedef __attribute__((ext_vector_type(8)))  float  v8f;

union BF16x16 { v16bf v; uint16_t u[16]; };
union F32x8   { v8f   v; float    f[8]; };

__device__ __forceinline__ uint16_t f2bf(float x) {
    union { float f; uint32_t u; } c; c.f = x;
    uint32_t b = c.u;
    return (uint16_t)((b + 0x7FFFu + ((b >> 16) & 1u)) >> 16);  // RNE
}

// ---------------- pack kernels ----------------
__global__ void pack_x_bf16(const float* __restrict__ x, uint16_t* __restrict__ xb, int n) {
    int i = blockIdx.x * blockDim.x + threadIdx.x;
    if (i < n) xb[i] = f2bf(x[i]);
}

// w_off (216,128,3,3) -> wb[224][1152], k = c*9 + (ky*3+kx), rows 216..223 zero
__global__ void pack_woff(const float* __restrict__ w, uint16_t* __restrict__ wb) {
    int i = blockIdx.x * blockDim.x + threadIdx.x;
    if (i >= CO1P * K1) return;
    int co = i / K1, k = i - co * K1;
    float v = 0.f;
    if (co < CO1) {
        int c = k / KTAP, r = k - c * KTAP;
        v = w[((size_t)co * CC + c) * KTAP + r];
    }
    wb[i] = f2bf(v);
}

__global__ void pack_wpw(const float* __restrict__ w, uint16_t* __restrict__ wb) {
    int i = blockIdx.x * blockDim.x + threadIdx.x;
    if (i < CC * CC) wb[i] = f2bf(w[i]);
}

// ---------------- conv1: block-tiled implicit GEMM via bf16 WMMA + LDS staging ----
// grid: B*32 blocks of 256 threads. Block = (batch b) x (128 spatial cols).
// Per K-step(32): stage A(224x32 weights) + B(32x128 im2col) in LDS; each wave
// owns one 16-col B fragment and sweeps all 14 M-tiles (14 WMMAs / wave / step).
__global__ void __launch_bounds__(256) conv1_wmma(const uint16_t* __restrict__ xb,
                                                  const uint16_t* __restrict__ wb,
                                                  float* __restrict__ t) {
    __shared__ uint16_t lA[CO1P * LPAD];   // 17920 B
    __shared__ uint16_t lB[128 * LPAD];    // 10240 B

    int b  = blockIdx.x >> 5;
    int n0 = (blockIdx.x & 31) * 128;      // spatial base (two image rows)
    int tid  = threadIdx.x;
    int w    = tid >> 5;
    int lane = tid & 31;
    int lm = lane & 15;
    int hl = lane >> 4;

    const uint16_t* xbb = xb + (size_t)b * CC * HWXY;
    int colw = n0 + w * 16 + lm;           // this wave's output column for lane

    v8f acc[14];
    #pragma unroll
    for (int mt = 0; mt < 14; ++mt) acc[mt] = {};

    for (int ks = 0; ks < K1; ks += 32) {
        // ---- stage A: 224 x 32 weights, coalesced ----
        #pragma unroll
        for (int e = 0; e < 28; ++e) {                  // 7168 / 256
            int idx = e * 256 + tid;
            int m = idx >> 5, kk = idx & 31;
            lA[m * LPAD + kk] = wb[(size_t)m * K1 + ks + kk];
        }
        if (ks + 32 < K1)                               // prefetch next weight tile
            __builtin_prefetch(wb + (size_t)(tid) * K1 + ks + 32, 0, 1);
        // ---- stage B: 32 x 128 im2col gather (shared by all 14 M-tiles) ----
        #pragma unroll
        for (int e = 0; e < 16; ++e) {                  // 4096 / 256
            int idx = e * 256 + tid;
            int K = idx >> 7, n = idx & 127;
            int Kg = ks + K;
            int c = Kg / KTAP;
            int r = Kg - c * KTAP;
            int ky = r / 3, kx = r - (r / 3) * 3;
            int col = n0 + n;
            int ii = col >> 6, jj = col & 63;
            int py = ii + ky - 1, px = jj + kx - 1;
            int pyc = py < 0 ? 0 : (py > 63 ? 63 : py);
            int pxc = px < 0 ? 0 : (px > 63 ? 63 : px);
            uint16_t raw = xbb[(size_t)c * HWXY + (pyc << 6) + pxc];
            lB[n * LPAD + K] = ((unsigned)py < 64u && (unsigned)px < 64u) ? raw : (uint16_t)0;
        }
        __syncthreads();

        // ---- B fragment for this wave (2x8 elems: k = 8h+q, 16+8h+q) ----
        BF16x16 bf;
        {
            const uint16_t* bp = &lB[(w * 16 + lm) * LPAD + hl * 8];
            #pragma unroll
            for (int q = 0; q < 8; ++q) { bf.u[q] = bp[q]; bf.u[8 + q] = bp[16 + q]; }
        }
        // ---- sweep all M tiles ----
        #pragma unroll
        for (int mt = 0; mt < 14; ++mt) {
            BF16x16 a;
            const uint16_t* ap = &lA[(mt * 16 + lm) * LPAD + hl * 8];
            #pragma unroll
            for (int q = 0; q < 8; ++q) { a.u[q] = ap[q]; a.u[8 + q] = ap[16 + q]; }
            acc[mt] = __builtin_amdgcn_wmma_f32_16x16x32_bf16(false, a.v, false, bf.v,
                                                              (short)0, acc[mt], false, false);
        }
        __syncthreads();
    }

    float* tb = t + (size_t)b * CO1P * HWXY;
    #pragma unroll
    for (int mt = 0; mt < 14; ++mt) {
        F32x8 res; res.v = acc[mt];
        int mbase = mt * 16 + hl * 8;
        #pragma unroll
        for (int r = 0; r < 8; ++r)
            tb[(size_t)(mbase + r) * HWXY + colw] = res.f[r];
    }
}

// ---------------- channel mean/max over 216 channels ----------------
__global__ void meanmax_k(const float* __restrict__ t, float* __restrict__ mm) {
    int tid = blockIdx.x * blockDim.x + threadIdx.x;          // 32768
    int b = tid >> 12, ij = tid & 4095;
    const float* tb = t + (size_t)b * CO1P * HWXY + ij;
    float s = 0.f, mx = -3.4e38f;
    for (int c = 0; c < CO1; ++c) {
        float v = tb[(size_t)c * HWXY];
        s += v; mx = v > mx ? v : mx;
    }
    mm[tid]         = s / (float)CO1;
    mm[32768 + tid] = mx;
}

// ---------------- 7x7 attention conv + sigmoid ----------------
__global__ void attn_k(const float* __restrict__ mm, const float* __restrict__ wsa,
                       float* __restrict__ attn) {
    int tid = blockIdx.x * blockDim.x + threadIdx.x;          // 32768
    int b = tid >> 12, ij = tid & 4095;
    int ii = ij >> 6, jj = ij & 63;
    float acc = 0.f;
    for (int d = 0; d < 2; ++d)
        for (int u = 0; u < 7; ++u) {
            int pi = ii + u - 3;
            if ((unsigned)pi >= 64u) continue;
            for (int v = 0; v < 7; ++v) {
                int pj = jj + v - 3;
                if ((unsigned)pj >= 64u) continue;
                acc += wsa[d * 49 + u * 7 + v] * mm[d * 32768 + (b << 12) + (pi << 6) + pj];
            }
        }
    attn[tid] = 1.f / (1.f + __expf(-acc));
}

// ---------------- deformable conv (gather-bound, fp32 VALU) + ReLU -> bf16 ----------------
__global__ void deform_k(const float* __restrict__ x, const float* __restrict__ t,
                         const float* __restrict__ attn, const float* __restrict__ wdef,
                         const float* __restrict__ bdef, uint16_t* __restrict__ featb) {
    int tid = blockIdx.x * blockDim.x + threadIdx.x;          // 32768
    int b = tid >> 12, ij = tid & 4095;
    int ii = ij >> 6, jj = ij & 63;
    const float* tb = t + (size_t)b * CO1P * HWXY + ij;
    const float* xg0 = x + (size_t)b * CC * HWXY;
    float at = attn[tid];

    for (int g = 0; g < GG; ++g) {
        float acc[CGRP];
        #pragma unroll
        for (int o = 0; o < CGRP; ++o) acc[o] = bdef[g * CGRP + o];

        for (int k = 0; k < KTAP; ++k) {
            int ky = k / 3, kx = k - ky * 3;
            int choff = (g * KTAP + k) * 2;
            float oy  = at * tb[(size_t)choff * HWXY];
            float ox  = at * tb[(size_t)(choff + 1) * HWXY];
            float msk = at * tb[(size_t)(2 * GG * KTAP + g * KTAP + k) * HWXY];
            msk = 1.f / (1.f + __expf(-msk));

            float py = oy + (float)(ii + ky - 1);
            float px = ox + (float)(jj + kx - 1);
            float y0f = floorf(py), x0f = floorf(px);
            float fy = py - y0f, fx = px - x0f;
            int y0 = (int)y0f, x0 = (int)x0f;
            int y1 = y0 + 1, x1 = x0 + 1;

            float w00 = (1.f - fy) * (1.f - fx) * msk;
            float w01 = (1.f - fy) * fx * msk;
            float w10 = fy * (1.f - fx) * msk;
            float w11 = fy * fx * msk;
            bool vy0 = (unsigned)y0 < 64u, vy1 = (unsigned)y1 < 64u;
            bool vx0 = (unsigned)x0 < 64u, vx1 = (unsigned)x1 < 64u;
            w00 = (vy0 && vx0) ? w00 : 0.f;
            w01 = (vy0 && vx1) ? w01 : 0.f;
            w10 = (vy1 && vx0) ? w10 : 0.f;
            w11 = (vy1 && vx1) ? w11 : 0.f;
            int y0c = y0 < 0 ? 0 : (y0 > 63 ? 63 : y0);
            int y1c = y1 < 0 ? 0 : (y1 > 63 ? 63 : y1);
            int x0c = x0 < 0 ? 0 : (x0 > 63 ? 63 : x0);
            int x1c = x1 < 0 ? 0 : (x1 > 63 ? 63 : x1);
            int i00 = (y0c << 6) + x0c, i01 = (y0c << 6) + x1c;
            int i10 = (y1c << 6) + x0c, i11 = (y1c << 6) + x1c;

            const float* xg  = xg0 + (size_t)g * CGRP * HWXY;
            const float* wdg = wdef + (size_t)g * CGRP * CGRP * KTAP; // [o][c][k]
            for (int c = 0; c < CGRP; ++c) {
                const float* xc = xg + (size_t)c * HWXY;
                float s = w00 * xc[i00] + w01 * xc[i01] + w10 * xc[i10] + w11 * xc[i11];
                #pragma unroll
                for (int o = 0; o < CGRP; ++o)
                    acc[o] += wdg[(size_t)o * (CGRP * KTAP) + c * KTAP + k] * s;
            }
        }
        #pragma unroll
        for (int o = 0; o < CGRP; ++o) {
            float v = acc[o] > 0.f ? acc[o] : 0.f;   // fused ReLU
            featb[((size_t)b * CC + g * CGRP + o) * HWXY + ij] = f2bf(v);
        }
    }
}

// ---------------- 1x1 conv: 128x128 GEMM via bf16 WMMA + LDS staging + bias ----
// grid: B*32 blocks of 256 threads. Block = (batch) x (128 cols); K=128 unrolled.
__global__ void __launch_bounds__(256) pw_wmma(const uint16_t* __restrict__ wb,
                                               const uint16_t* __restrict__ fb,
                                               const float* __restrict__ bias,
                                               float* __restrict__ out) {
    __shared__ uint16_t lA[CC * LPAD];     // 10240 B
    __shared__ uint16_t lB[128 * LPAD];    // 10240 B

    int b  = blockIdx.x >> 5;
    int n0 = (blockIdx.x & 31) * 128;
    int tid  = threadIdx.x;
    int w    = tid >> 5;
    int lane = tid & 31;
    int lm = lane & 15;
    int hl = lane >> 4;

    const uint16_t* fbb = fb + (size_t)b * CC * HWXY;
    int colw = n0 + w * 16 + lm;

    v8f acc[8];
    #pragma unroll
    for (int mt = 0; mt < 8; ++mt) acc[mt] = {};

    #pragma unroll
    for (int ks = 0; ks < CC; ks += 32) {
        #pragma unroll
        for (int e = 0; e < 16; ++e) {                  // A: 128x32
            int idx = e * 256 + tid;
            int m = idx >> 5, kk = idx & 31;
            lA[m * LPAD + kk] = wb[(size_t)m * CC + ks + kk];
        }
        #pragma unroll
        for (int e = 0; e < 16; ++e) {                  // B: 32x128, coalesced over n
            int idx = e * 256 + tid;
            int K = idx >> 7, n = idx & 127;
            lB[n * LPAD + K] = fbb[(size_t)(ks + K) * HWXY + n0 + n];
        }
        __syncthreads();

        BF16x16 bf;
        {
            const uint16_t* bp = &lB[(w * 16 + lm) * LPAD + hl * 8];
            #pragma unroll
            for (int q = 0; q < 8; ++q) { bf.u[q] = bp[q]; bf.u[8 + q] = bp[16 + q]; }
        }
        #pragma unroll
        for (int mt = 0; mt < 8; ++mt) {
            BF16x16 a;
            const uint16_t* ap = &lA[(mt * 16 + lm) * LPAD + hl * 8];
            #pragma unroll
            for (int q = 0; q < 8; ++q) { a.u[q] = ap[q]; a.u[8 + q] = ap[16 + q]; }
            acc[mt] = __builtin_amdgcn_wmma_f32_16x16x32_bf16(false, a.v, false, bf.v,
                                                              (short)0, acc[mt], false, false);
        }
        __syncthreads();
    }

    #pragma unroll
    for (int mt = 0; mt < 8; ++mt) {
        F32x8 res; res.v = acc[mt];
        int mbase = mt * 16 + hl * 8;
        #pragma unroll
        for (int r = 0; r < 8; ++r)
            out[((size_t)b * CC + mbase + r) * HWXY + colw] = res.f[r] + bias[mbase + r];
    }
}

// ---------------- launcher ----------------
extern "C" void kernel_launch(void* const* d_in, const int* in_sizes, int n_in,
                              void* d_out, int out_size, void* d_ws, size_t ws_size,
                              hipStream_t stream) {
    const float* x     = (const float*)d_in[0];
    const float* w_off = (const float*)d_in[1];
    const float* w_sa  = (const float*)d_in[2];
    const float* w_def = (const float*)d_in[3];
    const float* b_def = (const float*)d_in[4];
    const float* w_pw  = (const float*)d_in[5];
    const float* b_pw  = (const float*)d_in[6];
    float* out = (float*)d_out;

    char* ws = (char*)d_ws;
    uint16_t* xb    = (uint16_t*)(ws);                 //  8,388,608 B
    uint16_t* woffb = (uint16_t*)(ws +  8388608);      //    516,096 B
    float*    t     = (float*)   (ws +  8904704);      // 29,360,128 B
    float*    mm    = (float*)   (ws + 38264832);      //    262,144 B
    float*    attn  = (float*)   (ws + 38526976);      //    131,072 B
    uint16_t* featb = (uint16_t*)(ws + 38658048);      //  8,388,608 B
    uint16_t* wpwb  = (uint16_t*)(ws + 47046656);      //     32,768 B  (total ~47.1 MB)

    const int nX = BB * CC * HWXY;
    pack_x_bf16<<<(nX + 255) / 256, 256, 0, stream>>>(x, xb, nX);
    pack_woff  <<<(CO1P * K1 + 255) / 256, 256, 0, stream>>>(w_off, woffb);
    pack_wpw   <<<(CC * CC + 255) / 256, 256, 0, stream>>>(w_pw, wpwb);

    conv1_wmma <<<BB * 32, 256, 0, stream>>>(xb, woffb, t);      // 256 blocks
    meanmax_k  <<<128, 256, 0, stream>>>(t, mm);
    attn_k     <<<128, 256, 0, stream>>>(mm, w_sa, attn);
    deform_k   <<<128, 256, 0, stream>>>(x, t, attn, w_def, b_def, featb);
    pw_wmma    <<<BB * 32, 256, 0, stream>>>(wpwb, featb, b_pw, out);  // 256 blocks
}
